// RobustAugmentation_68418829025564
// MI455X (gfx1250) — matrix operand order repaired and checked by reading, not compiled, not run on hardware
//
#include <hip/hip_runtime.h>
#include <hip/hip_bf16.h>

#define B_ROWS 2048
#define M_MEM  131072
#define D_DIM  512
#define SPLIT  8
#define MCHUNK (M_MEM / SPLIT)   // 16384 keys per workgroup
#define MT     32                // keys per inner iteration
#define BT     32                // query rows per workgroup (2 row-groups of 16)
#define INV_T  (1.0f / 0.07f)
#define NEG_INF (-3.0e38f)

typedef __attribute__((ext_vector_type(16))) __bf16 v16bf;
typedef __attribute__((ext_vector_type(8)))  float  v8f;

// ---- CDNA5 async global->LDS copy (ASYNCcnt path), ISA 10. / 15.18.3 ----
// GV mode: per-lane 64-bit global address, per-lane 32-bit LDS byte address.
__device__ __forceinline__ void async_copy_b128(unsigned lds_addr, const void* gaddr) {
  asm volatile("global_load_async_to_lds_b128 %0, %1, off"
               :: "v"(lds_addr), "v"(gaddr)
               : "memory");
}
__device__ __forceinline__ void wait_asynccnt0() {
  asm volatile("s_wait_asynccnt 0x0" ::: "memory");
}

// ---- WMMA fragment loaders (contiguous k = 16*(lane/16) + s in both A and B, wave32) ----

// A (16x32 bf16) from global bf16: lane row = lane&15; K = k0..k0+15 contiguous, k0 = (lane<16)?0:16
__device__ __forceinline__ v16bf frag_a_from_bf16(const __bf16* __restrict__ base, int ld, int lane) {
  const int row = lane & 15;
  const int k0  = (lane < 16) ? 0 : 16;
  const __bf16* p = base + row * ld + k0;
  v16bf a;
#pragma unroll
  for (int j = 0; j < 8; ++j) a[j] = p[j];
#pragma unroll
  for (int j = 0; j < 8; ++j) a[8 + j] = p[8 + j];
  return a;
}

// B (32x16 bf16) from global fp32: lane column n = lane&15; K = (lane<16)?0..15:16..31 contiguous
__device__ __forceinline__ v16bf frag_b_from_f32(const float* __restrict__ base, int ld, int lane) {
  const int col = lane & 15;
  const int k0  = (lane < 16) ? 0 : 16;
  const float* p = base + col * ld + k0;
  v16bf b;
#pragma unroll
  for (int j = 0; j < 16; ++j) b[j] = (__bf16)p[j];
  return b;
}

// ---- Kernel 1: query = l2norm(x); emit fp32 (for final lerp) and bf16 (for WMMA A) ----
__global__ void __launch_bounds__(256) qnorm_kernel(const float* __restrict__ x,
                                                    float* __restrict__ q,
                                                    __bf16* __restrict__ qb) {
  __shared__ float red[256];
  const int b = blockIdx.x, tid = threadIdx.x;
  const float x0 = x[(size_t)b * D_DIM + tid];
  const float x1 = x[(size_t)b * D_DIM + tid + 256];
  red[tid] = x0 * x0 + x1 * x1;
  __syncthreads();
  for (int s = 128; s > 0; s >>= 1) { if (tid < s) red[tid] += red[tid + s]; __syncthreads(); }
  const float rn = rsqrtf(red[0]);
  const float q0 = x0 * rn, q1 = x1 * rn;
  q[(size_t)b * D_DIM + tid]        = q0;
  q[(size_t)b * D_DIM + tid + 256]  = q1;
  qb[(size_t)b * D_DIM + tid]       = (__bf16)q0;
  qb[(size_t)b * D_DIM + tid + 256] = (__bf16)q1;
}

// ---- Kernel 2: flash-attention partials; V staged by async DMA under score WMMAs ----
__global__ void __launch_bounds__(256, 1) flash_kernel(
    const __bf16* __restrict__ qb,    // [B][D] normalized queries (bf16)
    const float*  __restrict__ kmem,  // [M][D]
    const float*  __restrict__ vmem,  // [M][D]
    float* __restrict__ Opart,        // [SPLIT][B][D]
    float* __restrict__ mpart,        // [SPLIT][B]
    float* __restrict__ lpart)        // [SPLIT][B]
{
  __shared__ __align__(16) float VN[MT * D_DIM];   // V tile, natural [m][d] fp32, 64 KB (async dest)
  __shared__ __align__(32) __bf16 Pt[BT * MT];     // probs bf16 [row][m], 2 KB
  __shared__ float S0[BT * MT];                    // score partial, k-half 0 (4 KB)
  __shared__ float S1[BT * MT];                    // score partial, k-half 1 (4 KB)
  __shared__ float rowM[BT], rowL[BT], rowS[BT];

  const int tid   = threadIdx.x;
  const int lane  = tid & 31;
  const int wave  = tid >> 5;
  const int row0  = blockIdx.x * BT;
  const int chunk = blockIdx.y;
  const int mbase = chunk * MCHUNK;
  const int half8 = (lane >= 16) ? 8 : 0;

  // score-phase wave role: (row-group, m-tile, k-half)
  const int s_rg = (wave >> 2) & 1;
  const int s_mt = (wave >> 1) & 1;
  const int s_kh =  wave       & 1;
  // output-phase wave role: (row-group, 128-column quarter)
  const int o_rg = wave >> 2;
  const int o_d0 = (wave & 3) * 128;

  if (tid < BT) { rowM[tid] = NEG_INF; rowL[tid] = 0.0f; }

  v8f zero = {};
  v8f acc[8];
#pragma unroll
  for (int t = 0; t < 8; ++t) acc[t] = zero;

  const __bf16* qrow = qb + (size_t)(row0 + s_rg * 16) * D_DIM;
  const unsigned vn_lds = (unsigned)(size_t)(&VN[0]);   // LDS byte address (low 32 bits of flat)

  for (int it = 0; it < MCHUNK / MT; ++it) {
    const int m0 = mbase + it * MT;
    __syncthreads();  // previous iteration fully consumed VN/S/P

    __builtin_prefetch(kmem + (size_t)(m0 + MT) * D_DIM, 0, 1);

    // Kick off async DMA of the V tile: rows m0..m0+31 are one contiguous 64 KB
    // global block -> straight B128 async memcpy into LDS, overlapped with the
    // score WMMAs below. 16 x b128 per thread.
    {
      const char* gsrc = (const char*)(vmem + (size_t)m0 * D_DIM);
#pragma unroll
      for (int i = 0; i < 16; ++i) {
        const unsigned off = (unsigned)(tid + i * 256) * 16u;
        async_copy_b128(vn_lds + off, gsrc + off);
      }
    }

    // Scores: each wave computes a k-half partial of one 16x16 tile
    {
      v8f c = zero;
      const float* kbase = kmem + (size_t)(m0 + s_mt * 16) * D_DIM + s_kh * 256;
      const __bf16* abase = qrow + s_kh * 256;
#pragma unroll
      for (int kk = 0; kk < 8; ++kk) {
        v16bf a = frag_a_from_bf16(abase + kk * 32, D_DIM, lane);
        v16bf b = frag_b_from_f32(kbase + kk * 32, D_DIM, lane);
        c = __builtin_amdgcn_wmma_f32_16x16x32_bf16(false, a, false, b, (short)0, c, false, false);
      }
      float* Sbuf = s_kh ? S1 : S0;
      const int cn = s_mt * 16 + (lane & 15);
#pragma unroll
      for (int r = 0; r < 8; ++r)
        Sbuf[(s_rg * 16 + r + half8) * MT + cn] = c[r];
    }

    wait_asynccnt0();   // V tile DMA complete (this wave's asyncs)
    __syncthreads();    // ... and visible to all waves; scores ready too

    // Online softmax (one thread per query row); merge the two k-half partials here
    if (tid < BT) {
      const float mo = rowM[tid];
      float s[MT];
#pragma unroll 8
      for (int j = 0; j < MT; ++j) s[j] = (S0[tid * MT + j] + S1[tid * MT + j]) * INV_T;
      float cm = mo;
#pragma unroll 8
      for (int j = 0; j < MT; ++j) cm = fmaxf(cm, s[j]);
      const float sc = __expf(mo - cm);
      float sum = 0.0f;
#pragma unroll 8
      for (int j = 0; j < MT; ++j) {
        const float p = __expf(s[j] - cm);
        Pt[tid * MT + j] = (__bf16)p;
        sum += p;
      }
      rowL[tid] = rowL[tid] * sc + sum;
      rowM[tid] = cm;
      rowS[tid] = sc;
    }
    __syncthreads();

    // O(32x512) += P(32x32) * V(32x512); wave owns (row-group, 128 columns)
    float scl[8];
#pragma unroll
    for (int r = 0; r < 8; ++r) scl[r] = rowS[o_rg * 16 + r + half8];

    const int arow = lane & 15;
    const int ak0  = (lane < 16) ? 0 : 16;
    const __bf16* pp = Pt + (o_rg * 16 + arow) * MT + ak0;
    v16bf pa;
#pragma unroll
    for (int j = 0; j < 8; ++j) pa[j] = pp[j];
#pragma unroll
    for (int j = 0; j < 8; ++j) pa[8 + j] = pp[8 + j];

#pragma unroll
    for (int t = 0; t < 8; ++t) {
#pragma unroll
      for (int r = 0; r < 8; ++r) acc[t][r] *= scl[r];   // exp(m_old - m_new) rescale
      // B frag from VN (fp32, natural [m][d]): lane column d, 16 k's strided by D
      const int drow = o_d0 + t * 16 + (lane & 15);
      const float* vp = VN + ((lane < 16) ? 0 : 16) * D_DIM + drow;
      v16bf vb;
#pragma unroll
      for (int j = 0; j < 16; ++j) vb[j] = (__bf16)vp[j * D_DIM];
      acc[t] = __builtin_amdgcn_wmma_f32_16x16x32_bf16(false, pa, false, vb, (short)0, acc[t], false, false);
    }
  }

  // Write per-chunk partials
#pragma unroll
  for (int t = 0; t < 8; ++t) {
#pragma unroll
    for (int r = 0; r < 8; ++r) {
      const int row = o_rg * 16 + r + half8;
      const int col = o_d0 + t * 16 + (lane & 15);
      Opart[((size_t)chunk * B_ROWS + row0 + row) * D_DIM + col] = acc[t][r];
    }
  }
  if (tid < BT) {
    mpart[(size_t)chunk * B_ROWS + row0 + tid] = rowM[tid];
    lpart[(size_t)chunk * B_ROWS + row0 + tid] = rowL[tid];
  }
}

// ---- Kernel 3: merge splits (log-sum-exp), norms, interpolation ----
__global__ void __launch_bounds__(256) combine_kernel(
    const float* __restrict__ q,
    const float* __restrict__ Opart,
    const float* __restrict__ mpart,
    const float* __restrict__ lpart,
    float* __restrict__ out)
{
  __shared__ float red[256];
  const int b = blockIdx.x, tid = threadIdx.x;

  float gm = NEG_INF;
#pragma unroll
  for (int c = 0; c < SPLIT; ++c) gm = fmaxf(gm, mpart[(size_t)c * B_ROWS + b]);
  float L = 0.0f;
#pragma unroll
  for (int c = 0; c < SPLIT; ++c)
    L += lpart[(size_t)c * B_ROWS + b] * __expf(mpart[(size_t)c * B_ROWS + b] - gm);

  const int d0 = tid, d1 = tid + 256;
  float o0 = 0.0f, o1 = 0.0f;
#pragma unroll
  for (int c = 0; c < SPLIT; ++c) {
    const float w = __expf(mpart[(size_t)c * B_ROWS + b] - gm);
    const float* Ob = Opart + ((size_t)c * B_ROWS + b) * D_DIM;
    o0 += w * Ob[d0];
    o1 += w * Ob[d1];
  }
  const float invL = 1.0f / L;
  const float r0 = o0 * invL, r1 = o1 * invL;

  red[tid] = r0 * r0 + r1 * r1;
  __syncthreads();
  for (int s = 128; s > 0; s >>= 1) { if (tid < s) red[tid] += red[tid + s]; __syncthreads(); }
  const float rn = rsqrtf(red[0]);
  __syncthreads();

  const float q0 = q[(size_t)b * D_DIM + d0];
  const float q1 = q[(size_t)b * D_DIM + d1];
  const float v0 = 0.5f * q0 + 0.5f * r0 * rn;
  const float v1 = 0.5f * q1 + 0.5f * r1 * rn;

  red[tid] = v0 * v0 + v1 * v1;
  __syncthreads();
  for (int s = 128; s > 0; s >>= 1) { if (tid < s) red[tid] += red[tid + s]; __syncthreads(); }
  const float rn2 = rsqrtf(red[0]);

  out[(size_t)b * D_DIM + d0] = v0 * rn2;
  out[(size_t)b * D_DIM + d1] = v1 * rn2;
}

extern "C" void kernel_launch(void* const* d_in, const int* in_sizes, int n_in,
                              void* d_out, int out_size, void* d_ws, size_t ws_size,
                              hipStream_t stream) {
  (void)in_sizes; (void)n_in; (void)out_size; (void)ws_size;
  const float* x    = (const float*)d_in[0];
  const float* kmem = (const float*)d_in[1];
  const float* vmem = (const float*)d_in[2];
  float* out = (float*)d_out;
  float* ws  = (float*)d_ws;

  // Workspace layout: ~40 MB
  float*  qf32  = ws;                                          // [B][D] fp32
  float*  Opart = qf32  + (size_t)B_ROWS * D_DIM;              // [SPLIT][B][D]
  float*  mpart = Opart + (size_t)SPLIT * B_ROWS * D_DIM;      // [SPLIT][B]
  float*  lpart = mpart + (size_t)SPLIT * B_ROWS;              // [SPLIT][B]
  __bf16* qbf   = (__bf16*)(lpart + (size_t)SPLIT * B_ROWS);   // [B][D] bf16

  qnorm_kernel<<<B_ROWS, 256, 0, stream>>>(x, qf32, qbf);
  flash_kernel<<<dim3(B_ROWS / BT, SPLIT), 256, 0, stream>>>(qbf, kmem, vmem, Opart, mpart, lpart);
  combine_kernel<<<B_ROWS, 256, 0, stream>>>(qf32, Opart, mpart, lpart, out);
}
